// SinkhornOT_87479893885304
// MI455X (gfx1250) — matrix operand order repaired and checked by reading, not compiled, not run on hardware
//
#include <hip/hip_runtime.h>
#include <hip/hip_bf16.h>

typedef __attribute__((ext_vector_type(16))) _Float16 v16h;
typedef __attribute__((ext_vector_type(8)))  _Float16 v8h;
typedef __attribute__((ext_vector_type(8)))  float    v8f;

#define OT_EPS   0.05f
#define OT_ITERS 20
#define OT_MESH  3

// ---------- wave32 butterfly helpers (lanes 0-15 / 16-31 reduce independently) ----------
__device__ __forceinline__ float hmax16(float v) {
    v = fmaxf(v, __shfl_xor(v, 1, 32));
    v = fmaxf(v, __shfl_xor(v, 2, 32));
    v = fmaxf(v, __shfl_xor(v, 4, 32));
    v = fmaxf(v, __shfl_xor(v, 8, 32));
    return v;
}
__device__ __forceinline__ float hsum16(float v) {
    v += __shfl_xor(v, 1, 32);
    v += __shfl_xor(v, 2, 32);
    v += __shfl_xor(v, 4, 32);
    v += __shfl_xor(v, 8, 32);
    return v;
}
__device__ __forceinline__ float lse16(float x) {  // logsumexp over 16 lanes of a half-wave
    float m = hmax16(x);
    float s = hsum16(__expf(x - m));
    return m + __logf(s);
}

__device__ __forceinline__ v16h cat8(v8h lo, v8h hi) {
    v16h r;
#pragma unroll
    for (int i = 0; i < 8; ++i) { r[i] = lo[i]; r[8 + i] = hi[i]; }
    return r;
}

// ---------- one wave: TWO 16-row tiles through  y = act(x @ W^T + b) ----------
// Each 32x16 W B-tile is loaded once and fed to two WMMAs (q-tile and r-tile),
// halving global/L2 weight traffic vs. separate passes.
// Aq/Ar: LDS f16 16x256 row-major.  Wg: global f16 256x256 row-major.
__device__ __forceinline__ void mlp_layer_qr(const _Float16* Aq, const _Float16* Ar,
                                             const _Float16* __restrict__ Wg,
                                             const float* biasLds, bool relu,
                                             _Float16* Dq, _Float16* Dr, int lane) {
    const int n  = lane & 15;
    const int hi = lane >> 4;
#pragma unroll 1
    for (int nt = 0; nt < 16; ++nt) {
        v8f accq = {};
        v8f accr = {};
        const float bv = biasLds[nt * 16 + n];
        const _Float16* wrow = Wg + (size_t)(nt * 16 + n) * 256 + hi * 16;
#pragma unroll
        for (int ks = 0; ks < 8; ++ks) {
            const int kb = ks * 32;
            // B-matrix 32x16: lane (n, half=hi) holds B[hi*16+t][n] = W[nt*16+n][kb+hi*16+t]
            v16h bt = *(const v16h*)(wrow + kb);
            // A-matrix f16 16x32 layout: lane (m=n, half=hi): K[hi*16..+15]
            v16h aq = *(const v16h*)(Aq + n * 256 + kb + hi * 16);
            accq = __builtin_amdgcn_wmma_f32_16x16x32_f16(
                false, aq, false, bt, (short)0, accq, false, false);
            v16h ar = *(const v16h*)(Ar + n * 256 + kb + hi * 16);
            accr = __builtin_amdgcn_wmma_f32_16x16x32_f16(
                false, ar, false, bt, (short)0, accr, false, false);
        }
        // C/D layout: vgpr v -> row m = v + 8*hi, col = nt*16 + n
#pragma unroll
        for (int v = 0; v < 8; ++v) {
            float xq = accq[v] + bv;
            float xr = accr[v] + bv;
            if (relu) { xq = fmaxf(xq, 0.0f); xr = fmaxf(xr, 0.0f); }
            Dq[(v + 8 * hi) * 256 + nt * 16 + n] = (_Float16)xq;
            Dr[(v + 8 * hi) * 256 + nt * 16 + n] = (_Float16)xr;
        }
    }
}

// stage a 16x256 f32 tile -> LDS f16 row-major
__device__ __forceinline__ void stage_rows(const float* __restrict__ src,
                                           _Float16* dst, int lane) {
#pragma unroll 1
    for (int rr = 0; rr < 16; ++rr) {
        const float4 f0 = *(const float4*)(src + rr * 256 + lane * 8);
        const float4 f1 = *(const float4*)(src + rr * 256 + lane * 8 + 4);
        v8h h;
        h[0] = (_Float16)f0.x; h[1] = (_Float16)f0.y; h[2] = (_Float16)f0.z; h[3] = (_Float16)f0.w;
        h[4] = (_Float16)f1.x; h[5] = (_Float16)f1.y; h[6] = (_Float16)f1.z; h[7] = (_Float16)f1.w;
        *(v8h*)(dst + rr * 256 + lane * 8) = h;
    }
}

__global__ __launch_bounds__(256) void convert_w_kernel(const float* __restrict__ W1,
                                                        const float* __restrict__ W2,
                                                        _Float16* __restrict__ W1h,
                                                        _Float16* __restrict__ W2h) {
    int idx = blockIdx.x * 256 + threadIdx.x;
    if (idx < 256 * 256) {
        W1h[idx] = (_Float16)W1[idx];
        W2h[idx] = (_Float16)W2[idx];
    }
}

__global__ __launch_bounds__(64) void sinkhorn_fused_kernel(
    const float* __restrict__ slots_q, const float* __restrict__ slots_r,
    const float* __restrict__ mask_q,  const float* __restrict__ mask_r,
    const _Float16* __restrict__ W1h,  const float* __restrict__ b1,
    const _Float16* __restrict__ W2h,  const float* __restrict__ b2,
    float* __restrict__ o_sim, float* __restrict__ o_T,
    float* __restrict__ o_C,   float* __restrict__ o_cost)
{
    __shared__ _Float16 buf[2][4][16 * 256];   // per-wave quad buffers (64 KB total)
    __shared__ float biasA[256], biasB[256];
    __shared__ float scr[2][64];               // [0..15]=|q|^2 [16..31]=|r|^2 [32..47]=lmu [48..63]=lnu

    const int tid   = threadIdx.x;
    const int wave  = tid >> 5;
    const int lane  = tid & 31;
    const int batch = blockIdx.x * 2 + wave;
    const int n  = lane & 15;
    const int hi = lane >> 4;

    for (int i = tid; i < 256; i += 64) { biasA[i] = b1[i]; biasB[i] = b2[i]; }
    __syncthreads();

    _Float16* b0 = &buf[wave][0][0];
    _Float16* b1l = &buf[wave][1][0];
    _Float16* b2l = &buf[wave][2][0];
    _Float16* b3 = &buf[wave][3][0];

    // ---------- MLP projection: q and r fused through both layers ----------
    stage_rows(slots_q + (size_t)batch * 4096, b0, lane);
    stage_rows(slots_r + (size_t)batch * 4096, b1l, lane);
    __syncthreads();
    mlp_layer_qr(b0, b1l, W1h, biasA, true,  b2l, b3, lane);   // Hq->b2, Hr->b3
    __syncthreads();
    mlp_layer_qr(b2l, b3, W2h, biasB, false, b0, b1l, lane);   // Pq->b0, Pr->b1
    __syncthreads();

    // ---------- row squared norms ----------
    {
        const _Float16* P = hi ? b1l : b0;
        float s = 0.f;
#pragma unroll 1
        for (int d0 = 0; d0 < 256; d0 += 8) {
            v8h hv = *(const v8h*)(P + n * 256 + d0);
#pragma unroll
            for (int j = 0; j < 8; ++j) { float f = (float)hv[j]; s += f * f; }
        }
        scr[wave][hi * 16 + n] = s;
    }
    __syncthreads();

    // ---------- Gram: G[k][m] = Pq[k] . Pr[m], 8 WMMAs over K=256 ----------
    v8f G = {};
#pragma unroll
    for (int ks = 0; ks < 8; ++ks) {
        const int kb = ks * 32;
        v16h at = *(const v16h*)(b0 + n * 256 + kb + hi * 16);
        v16h bt = *(const v16h*)(b1l + n * 256 + kb + hi * 16);   // B[kk][m] = Pr[m][kb+kk]
        G = __builtin_amdgcn_wmma_f32_16x16x32_f16(
            false, at, false, bt, (short)0, G, false, false);
    }

    // element mapping: vgpr v -> k = v + 8*hi, column m = n
    float Cv[8];
    {
        const float nr_m = scr[wave][16 + n];
#pragma unroll
        for (int v = 0; v < 8; ++v) {
            float nq_k = scr[wave][v + 8 * hi];
            float d2 = nq_k + nr_m - 2.0f * G[v];
            Cv[v] = sqrtf(fmaxf(d2, 1e-6f));
            o_C[(size_t)batch * 256 + (v + 8 * hi) * 16 + n] = Cv[v];
        }
    }

    // ---------- normalized log marginals ----------
    {
        float mval = hi ? mask_r[batch * 16 + n] : mask_q[batch * 16 + n];
        float lm = __logf(fmaxf(mval, 1e-8f));
        scr[wave][32 + hi * 16 + n] = lm - lse16(lm);
    }
    __syncthreads();

    float lk[8];
    {
        const float lnu_m = scr[wave][48 + n];
#pragma unroll
        for (int v = 0; v < 8; ++v) {
            float lmu_k = scr[wave][32 + v + 8 * hi];
            lk[v] = -Cv[v] * (1.0f / OT_EPS) + lmu_k + lnu_m;
        }
    }

    // ---------- log-domain Sinkhorn ----------
    float la[8];
#pragma unroll
    for (int v = 0; v < 8; ++v) la[v] = 0.f;
    float lb = 0.f;
#pragma unroll 1
    for (int it = 0; it < OT_ITERS; ++it) {
#pragma unroll
        for (int v = 0; v < 8; ++v) la[v] = -lse16(lk[v] + lb);   // LSE over m
        float mx = -3.4e38f;                                       // LSE over k
#pragma unroll
        for (int v = 0; v < 8; ++v) mx = fmaxf(mx, lk[v] + la[v]);
        mx = fmaxf(mx, __shfl_xor(mx, 16, 32));
        float s = 0.f;
#pragma unroll
        for (int v = 0; v < 8; ++v) s += __expf(lk[v] + la[v] - mx);
        s += __shfl_xor(s, 16, 32);
        lb = -(mx + __logf(s));
    }

    float T[8];
#pragma unroll
    for (int v = 0; v < 8; ++v) T[v] = __expf(lk[v] + la[v] + lb);

    // ---------- MESH sharpening ----------
#pragma unroll 1
    for (int mi = 0; mi < OT_MESH; ++mi) {
#pragma unroll
        for (int v = 0; v < 8; ++v) T[v] *= T[v];
#pragma unroll
        for (int v = 0; v < 8; ++v) T[v] /= (hsum16(T[v]) + 1e-8f);   // row (over m)
        float cs = 0.f;
#pragma unroll
        for (int v = 0; v < 8; ++v) cs += T[v];
        cs += __shfl_xor(cs, 16, 32);                                  // col (over k)
#pragma unroll
        for (int v = 0; v < 8; ++v) T[v] /= (cs + 1e-8f);
    }

    // ---------- transport cost, similarity, outputs ----------
    float tc = 0.f;
#pragma unroll
    for (int v = 0; v < 8; ++v) tc += T[v] * Cv[v];
    tc = hsum16(tc);
    tc += __shfl_xor(tc, 16, 32);

#pragma unroll
    for (int v = 0; v < 8; ++v)
        o_T[(size_t)batch * 256 + (v + 8 * hi) * 16 + n] = T[v];
    if (lane == 0) {
        o_cost[batch] = tc;
        o_sim[batch]  = 1.0f / (1.0f + __expf(tc));   // sigmoid(-tc)
    }
}

extern "C" void kernel_launch(void* const* d_in, const int* in_sizes, int n_in,
                              void* d_out, int out_size, void* d_ws, size_t ws_size,
                              hipStream_t stream) {
    const float* slots_q = (const float*)d_in[0];
    const float* slots_r = (const float*)d_in[1];
    const float* mask_q  = (const float*)d_in[2];
    const float* mask_r  = (const float*)d_in[3];
    const float* W1      = (const float*)d_in[4];
    const float* b1      = (const float*)d_in[5];
    const float* W2      = (const float*)d_in[6];
    const float* b2      = (const float*)d_in[7];

    const int B = in_sizes[0] / (16 * 256);   // slots_q is [B,16,256]

    _Float16* W1h = (_Float16*)d_ws;
    _Float16* W2h = W1h + 256 * 256;

    convert_w_kernel<<<256, 256, 0, stream>>>(W1, W2, W1h, W2h);

    float* out    = (float*)d_out;
    float* o_sim  = out;                       // [B]
    float* o_T    = o_sim + B;                 // [B,16,16]
    float* o_C    = o_T + (size_t)B * 256;     // [B,16,16]
    float* o_cost = o_C + (size_t)B * 256;     // [B]

    sinkhorn_fused_kernel<<<B / 2, 64, 0, stream>>>(
        slots_q, slots_r, mask_q, mask_r, W1h, b1, W2h, b2,
        o_sim, o_T, o_C, o_cost);
}
